// TextPrototypeLayer_13486197309573
// MI455X (gfx1250) — compile-verified, matrix operand order checked by reading, not run on hardware
//
#include <hip/hip_runtime.h>
#include <hip/hip_bf16.h>

typedef __attribute__((ext_vector_type(16))) __bf16 v16bf;
typedef __attribute__((ext_vector_type(8)))  float  v8f;
typedef unsigned short u16;

#define DD   1024
#define VV   32000
#define ROWS 4096          // 16 * 256
#define NEG_INF (-1.0e30f)

__device__ __forceinline__ u16 f2bf_u(float f) {
    union { float f; unsigned u; } v; v.f = f;
    unsigned u = v.u;
    u += 0x7FFFu + ((u >> 16) & 1u);     // round-to-nearest-even
    return (u16)(u >> 16);
}
__device__ __forceinline__ __bf16 f2bf(float f) {
    u16 s = f2bf_u(f);
    return __builtin_bit_cast(__bf16, s);
}

// ---------------- prep: f32 -> bf16 convert ----------------
__global__ __launch_bounds__(256) void cvt_bf16(const float* __restrict__ src,
                                                u16* __restrict__ dst, int n) {
    for (int i = blockIdx.x * blockDim.x + threadIdx.x; i < n;
         i += gridDim.x * blockDim.x)
        dst[i] = f2bf_u(src[i]);
}

// ---------------- prep: T [V,D] f32 -> Tt [D,V] bf16 ----------------
__global__ __launch_bounds__(256) void transpose_bf16(const float* __restrict__ T,
                                                      u16* __restrict__ Tt) {
    __shared__ u16 tile[32][33];
    int d0 = blockIdx.x * 32;            // gridDim.x = DD/32
    int v0 = blockIdx.y * 32;            // gridDim.y = VV/32
    int tx = threadIdx.x & 31;
    int ty = threadIdx.x >> 5;           // 0..7
#pragma unroll
    for (int i = 0; i < 4; ++i) {
        int v = v0 + ty + i * 8;
        tile[ty + i * 8][tx] = f2bf_u(T[(size_t)v * DD + d0 + tx]);
    }
    __syncthreads();
#pragma unroll
    for (int i = 0; i < 4; ++i) {
        int d = d0 + ty + i * 8;
        Tt[(size_t)d * VV + v0 + tx] = tile[tx][ty + i * 8];
    }
}

// ---------------- pass 1: streaming log-sum-exp over V ----------------
__global__ __launch_bounds__(256) void pass1_lse(const u16* __restrict__ Xb,
                                                 const u16* __restrict__ Tb,
                                                 float* __restrict__ m_ws,
                                                 float* __restrict__ l_ws) {
    const int row0 = blockIdx.x * 16;
    const int lane = threadIdx.x & 31;
    const int wave = threadIdx.x >> 5;          // 0..7
    const int nlo  = lane & 15;
    const int hi   = lane >> 4;                 // 0/1

    // A operand resident: X tile 16x1024, 32 K-steps of 16 bf16 per lane
    v16bf xa[32];
    {
        const u16* xrow = Xb + (size_t)(row0 + nlo) * DD + hi * 16;
#pragma unroll
        for (int kk = 0; kk < 32; ++kk)
            xa[kk] = *(const v16bf*)(xrow + kk * 32);
    }

    float mloc[8], lloc[8];
#pragma unroll
    for (int r = 0; r < 8; ++r) { mloc[r] = NEG_INF; lloc[r] = 0.f; }

    // waves split V: wave w takes tiles vt = w*16, w*16+128, ...
    for (int vt = wave * 16; vt < VV; vt += 128) {
        const u16* trow = Tb + (size_t)(vt + nlo) * DD + hi * 16;
        __builtin_prefetch(trow + 128 * DD, 0, 1);   // next tile for this wave
        v8f c;
#pragma unroll
        for (int r = 0; r < 8; ++r) c[r] = 0.f;
#pragma unroll
        for (int kk = 0; kk < 32; ++kk) {
            v16bf bfrag = *(const v16bf*)(trow + kk * 32);
            c = __builtin_amdgcn_wmma_f32_16x16x32_bf16(
                    false, xa[kk], false, bfrag, (short)0, c, false, false);
        }
        // online softmax update; row m = r + 8*hi lives in a 16-lane half
#pragma unroll
        for (int r = 0; r < 8; ++r) {
            float s  = c[r];
            float mx = s;
            mx = fmaxf(mx, __shfl_xor(mx, 1, 16));
            mx = fmaxf(mx, __shfl_xor(mx, 2, 16));
            mx = fmaxf(mx, __shfl_xor(mx, 4, 16));
            mx = fmaxf(mx, __shfl_xor(mx, 8, 16));
            float mnew = fmaxf(mloc[r], mx);
            float p  = __expf(s - mnew);
            p += __shfl_xor(p, 1, 16);
            p += __shfl_xor(p, 2, 16);
            p += __shfl_xor(p, 4, 16);
            p += __shfl_xor(p, 8, 16);
            lloc[r] = lloc[r] * __expf(mloc[r] - mnew) + p;
            mloc[r] = mnew;
        }
    }

    // merge partial (m,l) across the 8 waves
    __shared__ float sm[8][16], sl[8][16];
    if (nlo == 0) {
#pragma unroll
        for (int r = 0; r < 8; ++r) {
            sm[wave][r + 8 * hi] = mloc[r];
            sl[wave][r + 8 * hi] = lloc[r];
        }
    }
    __syncthreads();
    if (threadIdx.x < 16) {
        float M = NEG_INF, L = 0.f;
#pragma unroll
        for (int w = 0; w < 8; ++w) {
            float mw = sm[w][threadIdx.x], lw = sl[w][threadIdx.x];
            float mn = fmaxf(M, mw);
            L = L * __expf(M - mn) + lw * __expf(mw - mn);
            M = mn;
        }
        m_ws[row0 + threadIdx.x] = M;
        l_ws[row0 + threadIdx.x] = L;
    }
}

// ---------------- pass 2: O = softmax(S) * T  (FA-2 split over D) ----------------
__global__ __launch_bounds__(256) void pass2_reprog(const u16* __restrict__ Xb,
                                                    const u16* __restrict__ Tb,
                                                    const u16* __restrict__ Tt,
                                                    const float* __restrict__ m_ws,
                                                    const float* __restrict__ l_ws,
                                                    float* __restrict__ O) {
    const int row0 = blockIdx.x * 16;
    const int lane = threadIdx.x & 31;
    const int wave = threadIdx.x >> 5;
    const int nlo  = lane & 15;
    const int hi   = lane >> 4;
    const int d0   = wave * 128;

    __shared__ u16 Pbuf[8][16][32];   // 8 KB: per-wave P tiles (row, vlocal) bf16

    v16bf xa[32];
    {
        const u16* xrow = Xb + (size_t)(row0 + nlo) * DD + hi * 16;
#pragma unroll
        for (int kk = 0; kk < 32; ++kk)
            xa[kk] = *(const v16bf*)(xrow + kk * 32);
    }

    float mrow[8], linv[8];
#pragma unroll
    for (int r = 0; r < 8; ++r) {
        int rr  = row0 + r + 8 * hi;
        mrow[r] = m_ws[rr];
        linv[r] = 1.f / l_ws[rr];
    }

    v8f acc[8];
#pragma unroll
    for (int t = 0; t < 8; ++t)
#pragma unroll
        for (int r = 0; r < 8; ++r) acc[t][r] = 0.f;

    for (int vb = 0; vb < VV; vb += 256) {
        // --- S stage: this wave's 32-v sub-chunk, two 16-col tiles ---
#pragma unroll
        for (int half = 0; half < 2; ++half) {
            int v0 = vb + wave * 32 + half * 16;
            const u16* trow = Tb + (size_t)(v0 + nlo) * DD + hi * 16;
            v8f c;
#pragma unroll
            for (int r = 0; r < 8; ++r) c[r] = 0.f;
#pragma unroll
            for (int kk = 0; kk < 32; ++kk) {
                v16bf bfrag = *(const v16bf*)(trow + kk * 32);
                c = __builtin_amdgcn_wmma_f32_16x16x32_bf16(
                        false, xa[kk], false, bfrag, (short)0, c, false, false);
            }
#pragma unroll
            for (int r = 0; r < 8; ++r) {
                float p = __expf(c[r] - mrow[r]);
                Pbuf[wave][r + 8 * hi][half * 16 + nlo] = f2bf_u(p);
            }
        }
        __syncthreads();
        // --- PV stage: all 8 P tiles against this wave's D slice ---
#pragma unroll
        for (int j = 0; j < 8; ++j) {
            v16bf pa = *(const v16bf*)(&Pbuf[j][nlo][hi * 16]);
            int vsrc = vb + j * 32;
#pragma unroll
            for (int t = 0; t < 8; ++t) {
                int dcol = d0 + t * 16 + nlo;
                v16bf bfrag = *(const v16bf*)(Tt + (size_t)dcol * VV + vsrc + hi * 16);
                acc[t] = __builtin_amdgcn_wmma_f32_16x16x32_bf16(
                             false, pa, false, bfrag, (short)0, acc[t], false, false);
            }
        }
        __syncthreads();
    }

#pragma unroll
    for (int t = 0; t < 8; ++t)
#pragma unroll
        for (int r = 0; r < 8; ++r)
            O[(size_t)(row0 + r + 8 * hi) * DD + d0 + t * 16 + nlo] = acc[t][r] * linv[r];
}

// ---------------- pass 3: E = O*W^T + b; LN(x + E) ----------------
__global__ __launch_bounds__(256) void pass3_out(const float* __restrict__ X,
                                                 const float* __restrict__ Obuf,
                                                 const u16* __restrict__ Wb,
                                                 const float* __restrict__ bias,
                                                 const float* __restrict__ gamma,
                                                 const float* __restrict__ beta,
                                                 float* __restrict__ out) {
    const int row0 = blockIdx.x * 16;
    const int lane = threadIdx.x & 31;
    const int wave = threadIdx.x >> 5;
    const int nlo  = lane & 15;
    const int hi   = lane >> 4;
    const int e0   = wave * 128;

    __shared__ float rsum[16], rsumsq[16];
    if (threadIdx.x < 16) { rsum[threadIdx.x] = 0.f; rsumsq[threadIdx.x] = 0.f; }
    __syncthreads();

    v8f acc[8];
#pragma unroll
    for (int t = 0; t < 8; ++t)
#pragma unroll
        for (int r = 0; r < 8; ++r) acc[t][r] = 0.f;

    for (int kk = 0; kk < 32; ++kk) {
        const float* orow = Obuf + (size_t)(row0 + nlo) * DD + kk * 32 + hi * 16;
        v16bf a;
#pragma unroll
        for (int i = 0; i < 16; ++i) a[i] = f2bf(orow[i]);
#pragma unroll
        for (int t = 0; t < 8; ++t) {
            const u16* wrow = Wb + (size_t)(e0 + t * 16 + nlo) * DD + kk * 32 + hi * 16;
            v16bf bfrag = *(const v16bf*)wrow;
            acc[t] = __builtin_amdgcn_wmma_f32_16x16x32_bf16(
                         false, a, false, bfrag, (short)0, acc[t], false, false);
        }
    }

    float psum[8], psq[8];
#pragma unroll
    for (int r = 0; r < 8; ++r) { psum[r] = 0.f; psq[r] = 0.f; }

#pragma unroll
    for (int t = 0; t < 8; ++t) {
        int e = e0 + t * 16 + nlo;
        float be = bias[e];
#pragma unroll
        for (int r = 0; r < 8; ++r) {
            int row = row0 + r + 8 * hi;
            float v = acc[t][r] + be + X[(size_t)row * DD + e];
            acc[t][r] = v;
            psum[r] += v;
            psq[r]  += v * v;
        }
    }
#pragma unroll
    for (int r = 0; r < 8; ++r) {
        psum[r] += __shfl_xor(psum[r], 1, 16);
        psum[r] += __shfl_xor(psum[r], 2, 16);
        psum[r] += __shfl_xor(psum[r], 4, 16);
        psum[r] += __shfl_xor(psum[r], 8, 16);
        psq[r]  += __shfl_xor(psq[r], 1, 16);
        psq[r]  += __shfl_xor(psq[r], 2, 16);
        psq[r]  += __shfl_xor(psq[r], 4, 16);
        psq[r]  += __shfl_xor(psq[r], 8, 16);
    }
    if (nlo == 0) {
#pragma unroll
        for (int r = 0; r < 8; ++r) {
            atomicAdd(&rsum[r + 8 * hi], psum[r]);
            atomicAdd(&rsumsq[r + 8 * hi], psq[r]);
        }
    }
    __syncthreads();

    const float invD = 1.0f / (float)DD;
#pragma unroll
    for (int r = 0; r < 8; ++r) {
        float mean = rsum[r + 8 * hi] * invD;
        float var  = rsumsq[r + 8 * hi] * invD - mean * mean;
        float rstd = rsqrtf(var + 1e-5f);
        int row = row0 + r + 8 * hi;
#pragma unroll
        for (int t = 0; t < 8; ++t) {
            int e = e0 + t * 16 + nlo;
            out[(size_t)row * DD + e] = (acc[t][r] - mean) * rstd * gamma[e] + beta[e];
        }
    }
}

// ---------------- host ----------------
extern "C" void kernel_launch(void* const* d_in, const int* in_sizes, int n_in,
                              void* d_out, int out_size, void* d_ws, size_t ws_size,
                              hipStream_t stream) {
    (void)in_sizes; (void)n_in; (void)out_size; (void)ws_size;
    const float* X     = (const float*)d_in[0];   // [16,256,1024]
    const float* T     = (const float*)d_in[1];   // [32000,1024]
    const float* W     = (const float*)d_in[2];   // [1024,1024]
    const float* bias  = (const float*)d_in[3];
    const float* gamma = (const float*)d_in[4];
    const float* beta  = (const float*)d_in[5];
    float* out = (float*)d_out;

    char* ws = (char*)d_ws;
    size_t off = 0;
    u16* Xb = (u16*)(ws + off); off += (size_t)ROWS * DD * 2;       // 8 MB
    u16* Tb = (u16*)(ws + off); off += (size_t)VV * DD * 2;         // 64 MB
    u16* Tt = (u16*)(ws + off); off += (size_t)VV * DD * 2;         // 64 MB
    u16* Wb = (u16*)(ws + off); off += (size_t)DD * DD * 2;         // 2 MB
    float* m_ws = (float*)(ws + off); off += (size_t)ROWS * 4;
    float* l_ws = (float*)(ws + off); off += (size_t)ROWS * 4;
    float* O    = (float*)(ws + off); off += (size_t)ROWS * DD * 4; // 16 MB

    cvt_bf16<<<2048, 256, 0, stream>>>(X, Xb, ROWS * DD);
    cvt_bf16<<<4096, 256, 0, stream>>>(T, Tb, VV * DD);
    cvt_bf16<<<1024, 256, 0, stream>>>(W, Wb, DD * DD);
    transpose_bf16<<<dim3(DD / 32, VV / 32), 256, 0, stream>>>(T, Tt);

    pass1_lse<<<ROWS / 16, 256, 0, stream>>>(Xb, Tb, m_ws, l_ws);
    pass2_reprog<<<ROWS / 16, 256, 0, stream>>>(Xb, Tb, Tt, m_ws, l_ws, O);
    pass3_out<<<ROWS / 16, 256, 0, stream>>>(X, O, Wb, bias, gamma, beta, out);
}
